// Recon_Only_Net_46231027974536
// MI455X (gfx1250) — matrix operand order repaired and checked by reading, not compiled, not run on hardware
//
#include <hip/hip_runtime.h>

#define NAa 1152
#define NDd 736
#define NPp 512
#define PADp 2048

typedef __attribute__((ext_vector_type(16))) _Float16 v16h;
typedef __attribute__((ext_vector_type(8)))  _Float16 v8h;
typedef __attribute__((ext_vector_type(8)))  float    v8f;

// ---- geometry constants (double-evaluated, stored as float) ----
constexpr double VOXd = 1.0 / 0.7;
constexpr double Dd   = VOXd * 595.0;          // source-isocenter
constexpr double DDd  = VOXd * 490.6;          // isocenter-detector
constexpr double DUd  = VOXd * 1.2858;         // detector pitch
constexpr double DSDd = Dd + DDd;
constexpr double PI_d = 3.14159265358979323846;
constexpr double Sd   = PI_d / (2.0 * NAa);    // pi/(2*NA) filter scale
// HU/window transform folded: out = fbp*C1 + C2, fbp = acc*Sd
constexpr double C1d  = Sd * 1000.0 / (0.0192 * 4096.0);
constexpr double C2d  = 24.0 / 4096.0;

// ============================================================
// Kernel 1: build spatial ramp kernel h[2048] + angle tables.
// Single workgroup; FILT half-spectrum staged in LDS between phases.
// ============================================================
__global__ void precompute_kernel(float* __restrict__ h,
                                  float* __restrict__ cosA,
                                  float* __restrict__ sinA) {
  __shared__ float FILT[PADp / 2 + 1];
  const int tid = threadIdx.x;
  const float PI = (float)PI_d;
  const float W0 = 2.0f * PI / (float)PADp;

  // Phase A: FILT[k] = 2*Re(fft(f))[k] * sinc window   (f even-symmetric)
  for (int k = tid; k <= PADp / 2; k += blockDim.x) {
    float s = 0.25f;                       // f[0]
    for (int mm = 0; mm < 512; ++mm) {
      int n = 2 * mm + 1;
      float pn = PI * (float)n;
      float fn = -1.0f / (pn * pn);        // f[n] (and f[2048-n])
      int t = (k * n) & (PADp - 1);        // exact range reduction
      s += 2.0f * fn * __cosf((float)t * W0);
    }
    float four = 2.0f * s;
    if (k > 0) {
      float w = PI * (float)k / (float)PADp;
      four *= __sinf(w) / w;               // Shepp-Logan window
    }
    FILT[k] = four;
  }
  __syncthreads();

  // Phase B: h = irfft(FILT, n=2048)
  for (int n = tid; n < PADp; n += blockDim.x) {
    float acc = FILT[0];
    for (int k = 1; k < PADp / 2; ++k) {
      int t = (k * n) & (PADp - 1);
      acc += 2.0f * FILT[k] * __cosf((float)t * W0);
    }
    acc += FILT[PADp / 2] * ((n & 1) ? -1.0f : 1.0f);
    h[n] = acc / (float)PADp;
  }

  // Angle tables: theta = 2*pi*a/NA + pi/2
  for (int a = tid; a < NAa; a += blockDim.x) {
    float th = 2.0f * PI * (float)a / (float)NAa + 0.5f * PI;
    cosA[a] = __cosf(th);
    sinA[a] = __sinf(th);
  }
}

// ============================================================
// Kernel 2: materialize convolution matrix H column-major, f16.
// Hcol[n*ND + k] = h[(n-k) mod 2048]  (B operand of the GEMM)
// ============================================================
__global__ void build_H_kernel(const float* __restrict__ h,
                               _Float16* __restrict__ Hcol) {
  int idx = blockIdx.x * blockDim.x + threadIdx.x;
  if (idx >= NDd * NDd) return;
  int n = idx / NDd;
  int k = idx - n * NDd;
  Hcol[idx] = (_Float16)h[(n - k) & (PADp - 1)];
}

// ============================================================
// Kernel 3: sinogram f32 -> f16
// ============================================================
__global__ void sino_cvt_kernel(const float* __restrict__ sino,
                                _Float16* __restrict__ s16) {
  int idx = blockIdx.x * blockDim.x + threadIdx.x;
  if (idx < NAa * NDd) s16[idx] = (_Float16)sino[idx];
}

// ============================================================
// Kernel 4: filtered = sino16 (1152x736) x Hcol (736x736), WMMA f16->f32.
// 8 waves / block, one 16x16 output tile per wave, K loop of 23x32.
// ============================================================
__global__ void wmma_filter_gemm(const _Float16* __restrict__ A,
                                 const _Float16* __restrict__ B,
                                 float* __restrict__ C) {
  const int lane = threadIdx.x & 31;
  const int wave = threadIdx.x >> 5;
  const int tile = blockIdx.x * 8 + wave;       // 72*46 = 3312 tiles = 414*8
  const int NT_N = NDd / 16;                    // 46
  const int tM = tile / NT_N;
  const int tN = tile - tM * NT_N;
  const int half = lane >> 4;                   // K-half selector
  const int m    = lane & 15;                   // row (A) / column (B,C)

  const _Float16* arow = A + (size_t)(tM * 16 + m) * NDd;  // A row-major
  const _Float16* bcol = B + (size_t)(tN * 16 + m) * NDd;  // H column-major

  v8f c = {0.f, 0.f, 0.f, 0.f, 0.f, 0.f, 0.f, 0.f};

  for (int kb = 0; kb < NDd; kb += 32) {
    // A fragment: lane holds K = kb + half*8 + {0..7} and +16
    v8h a0 = *(const v8h*)(arow + kb + half * 8);
    v8h a1 = *(const v8h*)(arow + kb + half * 8 + 16);
    v16h a = __builtin_shufflevector(a0, a1,
              0,1,2,3,4,5,6,7,8,9,10,11,12,13,14,15);
    // B fragment: lane holds column tN*16+m, K = kb + half*8 + {0..7} and +16
    v8h b0 = *(const v8h*)(bcol + kb + half * 8);
    v8h b1 = *(const v8h*)(bcol + kb + half * 8 + 16);
    v16h b = __builtin_shufflevector(b0, b1,
              0,1,2,3,4,5,6,7,8,9,10,11,12,13,14,15);
    c = __builtin_amdgcn_wmma_f32_16x16x32_f16(
            /*neg_a=*/false, a, /*neg_b=*/false, b,
            /*c_mod=*/(short)0, c, /*reuse_a=*/false, /*reuse_b=*/false);
  }

  // C/D layout: lane column n=m, rows r + 8*half in c[r]
  float* crow = C + (size_t)(tM * 16 + half * 8) * NDd + tN * 16 + m;
#pragma unroll
  for (int r = 0; r < 8; ++r) crow[(size_t)r * NDd] = c[r];
}

// ============================================================
// Kernel 5: fan-beam pixel-driven backprojection + HU window.
// 16x16 pixel tile per block; trig tables staged in LDS.
// ============================================================
__global__ void backproject_kernel(const float* __restrict__ F,
                                   const float* __restrict__ cosA,
                                   const float* __restrict__ sinA,
                                   float* __restrict__ out) {
  __shared__ float cbs[NAa];
  __shared__ float sbs[NAa];
  const int tid = threadIdx.y * 16 + threadIdx.x;
  for (int a = tid; a < NAa; a += 256) {
    cbs[a] = cosA[a];
    sbs[a] = sinA[a];
  }
  __syncthreads();

  const int ix = blockIdx.x * 16 + threadIdx.x;
  const int iy = blockIdx.y * 16 + threadIdx.y;
  const float x = (float)ix - (float)((NPp - 1) * 0.5);
  const float y = (float)iy - (float)((NPp - 1) * 0.5);

  const float Dc    = (float)Dd;
  const float DSDc  = (float)DSDd;
  const float invDU = (float)(1.0 / DUd);
  const float ctr   = (float)((NDd - 1) * 0.5);

  float acc = 0.0f;
  for (int a = 0; a < NAa; ++a) {
    float cb = cbs[a], sb = sbs[a];
    float pdot = x * cb + y * sb;       // along source direction
    float pe   = -x * sb + y * cb;      // along detector axis
    float den  = Dc - pdot;
    float inv  = 1.0f / den;
    float iu   = DSDc * pe * inv * invDU + ctr;
    float i0f  = floorf(iu);
    float frac = iu - i0f;
    int   i0i  = (int)i0f;
    bool valid = (i0i >= 0) & (i0i < NDd - 1) & (den > 0.0f);
    int   i0c  = min(max(i0i, 0), NDd - 2);
    const float* row = F + (size_t)a * NDd;
    float v = row[i0c] * (1.0f - frac) + row[i0c + 1] * frac;
    float wD = Dc * inv;
    acc += valid ? (wD * wD) * v : 0.0f;
  }
  out[(size_t)iy * NPp + ix] = acc * (float)C1d + (float)C2d;
}

// ============================================================
// launch
// ============================================================
extern "C" void kernel_launch(void* const* d_in, const int* in_sizes, int n_in,
                              void* d_out, int out_size, void* d_ws, size_t ws_size,
                              hipStream_t stream) {
  (void)in_sizes; (void)n_in; (void)out_size; (void)ws_size;
  const float* sino = (const float*)d_in[0];   // (1,1,1152,736) f32
  float* out = (float*)d_out;                  // (1,1,512,512)  f32
  char* ws = (char*)d_ws;

  // workspace layout (all offsets 16B aligned)
  float*    h     = (float*)(ws);                          //  2048 f32
  float*    cosA  = (float*)(ws + 8192);                   //  1152 f32
  float*    sinA  = (float*)(ws + 8192 + 4608);            //  1152 f32
  _Float16* s16   = (_Float16*)(ws + 17408);               //  NA*ND f16
  _Float16* Hcol  = (_Float16*)(ws + 17408 + 2 * NAa * NDd);
  float*    filt  = (float*)(ws + 17408 + 2 * NAa * NDd + 2 * NDd * NDd);

  precompute_kernel<<<1, 1024, 0, stream>>>(h, cosA, sinA);
  build_H_kernel<<<(NDd * NDd + 255) / 256, 256, 0, stream>>>(h, Hcol);
  sino_cvt_kernel<<<(NAa * NDd + 255) / 256, 256, 0, stream>>>(sino, s16);
  wmma_filter_gemm<<<(NAa / 16) * (NDd / 16) / 8, 256, 0, stream>>>(s16, Hcol, filt);
  backproject_kernel<<<dim3(NPp / 16, NPp / 16), dim3(16, 16), 0, stream>>>(
      filt, cosA, sinA, out);
}
